// BiLevelRoutingAttention_42082089566405
// MI455X (gfx1250) — compile-verified, matrix operand order checked by reading, not run on hardware
//
#include <hip/hip_runtime.h>
#include <hip/hip_bf16.h>

// ---------------------------------------------------------------------------
// BiLevelRoutingAttention for MI455X (gfx1250, wave32, WMMA 16x16x32 bf16)
// N=8, H=W=56, DIM=QK_DIM=512, 16 heads x 32, 7x7 windows of 8x8=64 px, TOPK=4
// ---------------------------------------------------------------------------

typedef __attribute__((ext_vector_type(16))) __bf16 v16bf;
typedef __attribute__((ext_vector_type(8)))  float  v8f;
typedef __attribute__((ext_vector_type(4)))  float  f32x4;

union AFrag { v16bf v; uint4 u[2]; __bf16 e[16]; };
union CFrag { v8f   v; float  e[8];  };
union Pack16 { __bf16 h[16]; uint4 u[2]; };
union Pack8  { __bf16 h[8];  uint4 u;    };

__device__ constexpr float kScale = 0.044194173824159216f; // 512^-0.5

// ---------------------------------------------------------------------------
// K1/K6: C[M x Nc] = A[M x K] @ W[Nc x K]^T + bias   (bf16 WMMA, fp32 accum)
// mode 0: rows of C window-ordered, rows of A image-ordered (fused partition)
// mode 1: identity row mapping.
// grid (M/128, Nc/128), block 256 (8 waves), dyn LDS = 16 KB
// ---------------------------------------------------------------------------
__global__ void brt_gemm_bf16(const float* __restrict__ A,
                              const float* __restrict__ W,
                              const float* __restrict__ bias,
                              float* __restrict__ C,
                              int M, int Nc, int K, int mode)
{
  extern __shared__ char smem[];
  __bf16* sA = (__bf16*)smem;                   // [128][32]
  __bf16* sW = (__bf16*)(smem + 128 * 32 * 2);  // [128][32]

  const int tid  = threadIdx.x;
  const int lane = tid & 31;
  const int wave = tid >> 5;
  const int mblk = blockIdx.x;
  const int nblk = blockIdx.y;

  const int mg = wave & 3;   // M-tiles {mg*2, mg*2+1}
  const int ng = wave >> 2;  // N-tiles [ng*4, ng*4+4)

  // staging: 2 threads per row, 16 contiguous channels each
  const int srow = tid >> 1;         // local row 0..127
  const int sh   = (tid & 1) * 16;   // channel base within the 32-chunk

  // source row for A (window permute fused), loop-invariant
  long long arow;
  {
    int grow = mblk * 128 + srow;
    if (mode == 0) {
      int n   = grow / (49 * 64);
      int rem = grow - n * 49 * 64;
      int p   = rem >> 6, pix = rem & 63;
      arow = (long long)(n * 56 + (p / 7) * 8 + (pix >> 3)) * 56 +
             (p % 7) * 8 + (pix & 7);
    } else {
      arow = grow;
    }
  }
  const long long wrow = (long long)nblk * 128 + srow;

  CFrag acc[2][4];
#pragma unroll
  for (int mi = 0; mi < 2; ++mi)
#pragma unroll
    for (int t = 0; t < 4; ++t) { v8f z = {}; acc[mi][t].v = z; }

  const int kb = (lane >> 4) * 8;

  for (int k0 = 0; k0 < K; k0 += 32) {
    // --- stage A tile (16 floats/thread, vectorized) ---
    {
      const float* src = A + arow * K + k0 + sh;
      f32x4 f0 = ((const f32x4*)src)[0];
      f32x4 f1 = ((const f32x4*)src)[1];
      f32x4 f2 = ((const f32x4*)src)[2];
      f32x4 f3 = ((const f32x4*)src)[3];
      Pack16 pk;
#pragma unroll
      for (int j = 0; j < 4; ++j) {
        pk.h[j]      = (__bf16)f0[j];
        pk.h[4 + j]  = (__bf16)f1[j];
        pk.h[8 + j]  = (__bf16)f2[j];
        pk.h[12 + j] = (__bf16)f3[j];
      }
      uint4* dst = (uint4*)&sA[srow * 32 + sh];
      dst[0] = pk.u[0];
      dst[1] = pk.u[1];
    }
    // --- stage W slab ---
    {
      const float* src = W + wrow * K + k0 + sh;
      f32x4 f0 = ((const f32x4*)src)[0];
      f32x4 f1 = ((const f32x4*)src)[1];
      f32x4 f2 = ((const f32x4*)src)[2];
      f32x4 f3 = ((const f32x4*)src)[3];
      Pack16 pk;
#pragma unroll
      for (int j = 0; j < 4; ++j) {
        pk.h[j]      = (__bf16)f0[j];
        pk.h[4 + j]  = (__bf16)f1[j];
        pk.h[8 + j]  = (__bf16)f2[j];
        pk.h[12 + j] = (__bf16)f3[j];
      }
      uint4* dst = (uint4*)&sW[srow * 32 + sh];
      dst[0] = pk.u[0];
      dst[1] = pk.u[1];
    }
    __syncthreads();

    // --- fragments + 8 WMMAs ---
    AFrag a[2];
#pragma unroll
    for (int mi = 0; mi < 2; ++mi) {
      int r = (mg * 2 + mi) * 16 + (lane & 15);
      a[mi].u[0] = *(const uint4*)&sA[r * 32 + kb];
      a[mi].u[1] = *(const uint4*)&sA[r * 32 + 16 + kb];
    }
#pragma unroll
    for (int t = 0; t < 4; ++t) {
      int ncol = (ng * 4 + t) * 16 + (lane & 15);
      AFrag b;
      b.u[0] = *(const uint4*)&sW[ncol * 32 + kb];
      b.u[1] = *(const uint4*)&sW[ncol * 32 + 16 + kb];
#pragma unroll
      for (int mi = 0; mi < 2; ++mi) {
        acc[mi][t].v = __builtin_amdgcn_wmma_f32_16x16x32_bf16(
            false, a[mi].v, false, b.v, (short)0, acc[mi][t].v, false, false);
      }
    }
    __syncthreads();
  }

#pragma unroll
  for (int t = 0; t < 4; ++t) {
    int col = nblk * 128 + (ng * 4 + t) * 16 + (lane & 15);
    float bv = bias[col];
#pragma unroll
    for (int mi = 0; mi < 2; ++mi) {
#pragma unroll
      for (int v = 0; v < 8; ++v) {
        int row = mblk * 128 + (mg * 2 + mi) * 16 + v + ((lane >> 4) << 3);
        C[(long long)row * Nc + col] = acc[mi][t].e[v] + bv;
      }
    }
  }
}

// ---------------------------------------------------------------------------
// K2: per-window means of q (ch 0..511) and k (ch 512..1023); grid N*49
// ---------------------------------------------------------------------------
__global__ void brt_win_means(const float* __restrict__ qkv,
                              float* __restrict__ q_win,
                              float* __restrict__ k_win)
{
  int np = blockIdx.x;
  for (int c = threadIdx.x; c < 1024; c += blockDim.x) {
    const float* base = qkv + (long long)np * 64 * 1536 + c;
    float s = 0.f;
    for (int t = 0; t < 64; ++t) s += base[(long long)t * 1536];
    s *= (1.0f / 64.0f);
    if (c < 512) q_win[np * 512 + c] = s;
    else         k_win[np * 512 + (c - 512)] = s;
  }
}

// ---------------------------------------------------------------------------
// K3: router logits (49x49 per image) + top-4 indices; grid N, block 64
// ---------------------------------------------------------------------------
__global__ void brt_router_topk(const float* __restrict__ q_win,
                                const float* __restrict__ k_win,
                                int* __restrict__ r_idx)
{
  int n = blockIdx.x;
  int p = threadIdx.x;
  if (p >= 49) return;
  const float* q = q_win + (long long)(n * 49 + p) * 512;
  float best[4] = {-1e30f, -1e30f, -1e30f, -1e30f};
  int   bidx[4] = {0, 0, 0, 0};
  for (int qq = 0; qq < 49; ++qq) {
    const float* k = k_win + (long long)(n * 49 + qq) * 512;
    float d = 0.f;
    for (int c = 0; c < 512; ++c) d += q[c] * k[c];
    d *= kScale;
    for (int s = 0; s < 4; ++s) {
      if (d > best[s]) {
        for (int u = 3; u > s; --u) { best[u] = best[u - 1]; bidx[u] = bidx[u - 1]; }
        best[s] = d; bidx[s] = qq;
        break;
      }
    }
  }
  for (int s = 0; s < 4; ++s) r_idx[(n * 49 + p) * 4 + s] = bidx[s];
}

// ---------------------------------------------------------------------------
// K4: gathered window attention; one block per (n, window, head); 4 waves.
// S = Q*scale @ K^T (16 WMMA), in-LDS softmax, O = P @ V (16 WMMA).
// V staged TRANSPOSED so O-stage B-fragments are contiguous ds_load_b128.
// dyn LDS 64 KB: sK[256][32] + sVt[32][256] + sS[64][256] (all bf16)
// ---------------------------------------------------------------------------
__global__ void brt_window_attn(const float* __restrict__ qkv,
                                const int* __restrict__ r_idx,
                                float* __restrict__ out)
{
  int b    = blockIdx.x;
  int head = b & 15;
  int np   = b >> 4;
  int n = np / 49, p = np % 49;

  const int tid = threadIdx.x, lane = tid & 31, wave = tid >> 5;

  extern __shared__ char smem[];
  __bf16* sK  = (__bf16*)smem;                      // [256][32] row-major
  __bf16* sVt = (__bf16*)(smem + 256 * 32 * 2);     // [32][256] transposed
  __bf16* sS  = (__bf16*)(smem + 2 * 256 * 32 * 2); // [64][256]

  int tps[4];
#pragma unroll
  for (int s = 0; s < 4; ++s) tps[s] = r_idx[(n * 49 + p) * 4 + s];

  // gather K (row-major) and V (transposed) for the 4 routed windows
  for (int i = tid; i < 1024; i += 128) {   // 256 px * 4 chunks of 8 ch
    int t  = i >> 2;
    int cg = (i & 3) * 8;
    int s = t >> 6, pw = t & 63;
    long long rb =
        ((long long)(n * 49 + tps[s]) * 64 + pw) * 1536 + head * 32 + cg;
    const float* kp = qkv + rb + 512;
    const float* vp = qkv + rb + 1024;
    f32x4 k0 = ((const f32x4*)kp)[0];
    f32x4 k1 = ((const f32x4*)kp)[1];
    Pack8 pk;
#pragma unroll
    for (int j = 0; j < 4; ++j) {
      pk.h[j]     = (__bf16)k0[j];
      pk.h[4 + j] = (__bf16)k1[j];
    }
    *(uint4*)&sK[t * 32 + cg] = pk.u;
    f32x4 v0 = ((const f32x4*)vp)[0];
    f32x4 v1 = ((const f32x4*)vp)[1];
#pragma unroll
    for (int j = 0; j < 4; ++j) {
      sVt[(cg + j) * 256 + t]     = (__bf16)v0[j];
      sVt[(cg + 4 + j) * 256 + t] = (__bf16)v1[j];
    }
  }
  __syncthreads();

  const int kb = (lane >> 4) * 8;

  // Q fragment (scale folded); head dim 32 == one WMMA K-step
  AFrag aq;
  {
    int qrow = wave * 16 + (lane & 15);
    const float* qp = qkv + ((long long)np * 64 + qrow) * 1536 + head * 32;
    f32x4 q0 = ((const f32x4*)(qp + kb))[0];
    f32x4 q1 = ((const f32x4*)(qp + kb))[1];
    f32x4 q2 = ((const f32x4*)(qp + 16 + kb))[0];
    f32x4 q3 = ((const f32x4*)(qp + 16 + kb))[1];
#pragma unroll
    for (int j = 0; j < 4; ++j) {
      aq.e[j]      = (__bf16)(q0[j] * kScale);
      aq.e[4 + j]  = (__bf16)(q1[j] * kScale);
      aq.e[8 + j]  = (__bf16)(q2[j] * kScale);
      aq.e[12 + j] = (__bf16)(q3[j] * kScale);
    }
  }

  // S = Q K^T : 16 N-tiles of 16
  for (int nt = 0; nt < 16; ++nt) {
    AFrag bk;
    int col = nt * 16 + (lane & 15);
    bk.u[0] = *(const uint4*)&sK[col * 32 + kb];
    bk.u[1] = *(const uint4*)&sK[col * 32 + 16 + kb];
    v8f z = {};
    CFrag s;
    s.v = __builtin_amdgcn_wmma_f32_16x16x32_bf16(
        false, aq.v, false, bk.v, (short)0, z, false, false);
#pragma unroll
    for (int v = 0; v < 8; ++v) {
      int row = wave * 16 + v + ((lane >> 4) << 3);
      sS[row * 256 + col] = (__bf16)s.e[v];
    }
  }
  __syncthreads();

  // row softmax over 256 keys (one thread per row)
  if (tid < 64) {
    float mx = -1e30f;
    for (int c = 0; c < 256; ++c) {
      float x = (float)sS[tid * 256 + c];
      mx = fmaxf(mx, x);
    }
    float sum = 0.f;
    for (int c = 0; c < 256; ++c) sum += __expf((float)sS[tid * 256 + c] - mx);
    float inv = 1.0f / sum;
    for (int c = 0; c < 256; ++c)
      sS[tid * 256 + c] = (__bf16)(__expf((float)sS[tid * 256 + c] - mx) * inv);
  }
  __syncthreads();

  // O = P @ V : 2 N-tiles of 16, K loop 8 x 32 (all fragments b128-contiguous)
  CFrag o[2];
  { v8f z = {}; o[0].v = z; o[1].v = z; }
  for (int kt = 0; kt < 8; ++kt) {
    AFrag ap;
    int prow = wave * 16 + (lane & 15);
    ap.u[0] = *(const uint4*)&sS[prow * 256 + kt * 32 + kb];
    ap.u[1] = *(const uint4*)&sS[prow * 256 + kt * 32 + 16 + kb];
#pragma unroll
    for (int nt = 0; nt < 2; ++nt) {
      AFrag bv;
      int c = nt * 16 + (lane & 15);
      bv.u[0] = *(const uint4*)&sVt[c * 256 + kt * 32 + kb];
      bv.u[1] = *(const uint4*)&sVt[c * 256 + kt * 32 + 16 + kb];
      o[nt].v = __builtin_amdgcn_wmma_f32_16x16x32_bf16(
          false, ap.v, false, bv.v, (short)0, o[nt].v, false, false);
    }
  }

  // scatter to image layout (undo window partition)
  int jj = p / 7, ii = p % 7;
#pragma unroll
  for (int nt = 0; nt < 2; ++nt) {
    int ch = head * 32 + nt * 16 + (lane & 15);
#pragma unroll
    for (int v = 0; v < 8; ++v) {
      int q = wave * 16 + v + ((lane >> 4) << 3);
      int y = jj * 8 + (q >> 3), x = ii * 8 + (q & 7);
      out[(((long long)n * 56 + y) * 56 + x) * 512 + ch] = o[nt].e[v];
    }
  }
}

// ---------------------------------------------------------------------------
// K5: depthwise 3x3 lepe conv on V (read from window-ordered qkv) + bias,
// accumulated into attn_out (image layout). One thread per output element.
// ---------------------------------------------------------------------------
__global__ void brt_lepe_add(const float* __restrict__ qkv,
                             const float* __restrict__ w,
                             const float* __restrict__ bias,
                             float* __restrict__ out)
{
  long long idx = (long long)blockIdx.x * blockDim.x + threadIdx.x;
  const long long total = (long long)8 * 56 * 56 * 512;
  if (idx >= total) return;
  int c = (int)(idx & 511);
  long long pix = idx >> 9;
  int x = (int)(pix % 56);
  long long t = pix / 56;
  int y = (int)(t % 56);
  int n = (int)(t / 56);

  float s = bias[c];
#pragma unroll
  for (int ky = -1; ky <= 1; ++ky) {
    int yy = y + ky;
    if (yy < 0 || yy > 55) continue;
#pragma unroll
    for (int kx = -1; kx <= 1; ++kx) {
      int xx = x + kx;
      if (xx < 0 || xx > 55) continue;
      int jj = yy >> 3, hh = yy & 7, ii = xx >> 3, ww = xx & 7;
      long long off =
          ((long long)(n * 49 + jj * 7 + ii) * 64 + hh * 8 + ww) * 1536 + 1024 + c;
      s += qkv[off] * w[c * 9 + (ky + 1) * 3 + (kx + 1)];
    }
  }
  out[idx] += s;
}

// ---------------------------------------------------------------------------
// launch
// ---------------------------------------------------------------------------
extern "C" void kernel_launch(void* const* d_in, const int* in_sizes, int n_in,
                              void* d_out, int out_size, void* d_ws, size_t ws_size,
                              hipStream_t stream)
{
  const float* x      = (const float*)d_in[0];
  const float* qkv_w  = (const float*)d_in[1];
  const float* qkv_b  = (const float*)d_in[2];
  const float* wo_w   = (const float*)d_in[3];
  const float* wo_b   = (const float*)d_in[4];
  const float* lepe_w = (const float*)d_in[5];
  const float* lepe_b = (const float*)d_in[6];
  float* outp = (float*)d_out;

  const int M = 8 * 49 * 64;                 // 25088 rows
  const size_t qkv_elems  = (size_t)M * 1536;
  const size_t attn_elems = (size_t)8 * 56 * 56 * 512;
  const size_t win_elems  = (size_t)8 * 49 * 512;

  char* ws = (char*)d_ws;
  float* qkv_buf  = (float*)ws;                       ws += qkv_elems  * 4;
  float* attn_out = (float*)ws;                       ws += attn_elems * 4;
  float* q_win    = (float*)ws;                       ws += win_elems  * 4;
  float* k_win    = (float*)ws;                       ws += win_elems  * 4;
  int*   r_idx    = (int*)ws;

  const size_t gemm_lds = (size_t)(2 * 128 * 32) * sizeof(__bf16);          // 16 KB
  const size_t attn_lds = (size_t)(2 * 256 * 32 + 64 * 256) * sizeof(__bf16); // 64 KB

  // 1) fused window-partition + QKV projection (WMMA bf16)
  brt_gemm_bf16<<<dim3(M / 128, 1536 / 128), 256, gemm_lds, stream>>>(
      x, qkv_w, qkv_b, qkv_buf, M, 1536, 512, /*mode=*/0);

  // 2) window means for router
  brt_win_means<<<8 * 49, 256, 0, stream>>>(qkv_buf, q_win, k_win);

  // 3) router logits + top-4
  brt_router_topk<<<8, 64, 0, stream>>>(q_win, k_win, r_idx);

  // 4) gathered window attention (WMMA bf16), scatters to image layout
  brt_window_attn<<<8 * 49 * 16, 128, attn_lds, stream>>>(
      qkv_buf, r_idx, attn_out);

  // 5) lepe depthwise conv accumulate
  {
    long long total = (long long)attn_elems;
    int blocks = (int)((total + 255) / 256);
    brt_lepe_add<<<blocks, 256, 0, stream>>>(qkv_buf, lepe_w, lepe_b, attn_out);
  }

  // 6) output projection (WMMA bf16) -> d_out
  brt_gemm_bf16<<<dim3(M / 128, 512 / 128), 256, gemm_lds, stream>>>(
      attn_out, wo_w, wo_b, outp, M, 512, 512, /*mode=*/1);
}
